// ProtoPNetConv_53102975648265
// MI455X (gfx1250) — matrix operand order repaired and checked by reading, not compiled, not run on hardware
//
#include <hip/hip_runtime.h>
#include <hip/hip_bf16.h>

typedef __attribute__((ext_vector_type(16))) __bf16 v16bf;
typedef __attribute__((ext_vector_type(8)))  float  v8f;

// Problem constants
#define B_    64
#define L_    256
#define E_    1600
#define P_    100
#define M_    (B_ * L_)      // 16384 rows of GEMM
#define N_    (P_ * 4)       // 400 cols  (p*4 + tap)
#define KST   (E_ / 32)      // 50 k-steps of 32
#define NT    (N_ / 16)      // 25 column tiles per wave
#define CHUNK4  1600         // valid uint4 per 32-deep W chunk (25.6 KB)
#define CHUNKP4 1664         // padded chunk stride in uint4 (13 * 128)

// Output layout (floats)
#define D1LOUT 253
#define D2LOUT 250
#define D3LOUT 247
#define OFF_D1 6400
#define OFF_D2 (OFF_D1 + B_*34*D1LOUT)   // 6400 + 550528
#define OFF_D3 (OFF_D2 + B_*33*D2LOUT)   // + 528000
#define OFF_CL (OFF_D3 + B_*33*D3LOUT)   // + 521664 -> 1606592

union Frag { uint4 u[2]; v16bf v; };

// ---- CDNA5 async global->LDS copy (ASYNCcnt path) --------------------------
// Input-only operands + "memory" clobber (this exact form assembled in an
// earlier round; "=m" LDS operands crash AMDGPU isel). The VDST operand is the
// wave-relative LDS byte address; ldsW is the only static LDS object in
// proto_gemm, so its base offset is 0.
__device__ __forceinline__ void async_g2l_b128(unsigned ldsByteOff, const uint4* g) {
    asm volatile("global_load_async_to_lds_b128 %0, %1, off"
                 :: "v"(ldsByteOff), "v"(g) : "memory");
}
__device__ __forceinline__ void wait_async0() {
#if defined(__has_builtin) && __has_builtin(__builtin_amdgcn_s_wait_asynccnt)
    __builtin_amdgcn_s_wait_asynccnt(0);
#else
    asm volatile("s_wait_asynccnt 0x0" ::: "memory");
#endif
}

// ---------------------------------------------------------------------------
// Pack embedding [16384,1600] f32 -> bf16 A-fragments.
// Apack[((rgroup*KST + ks)*32 + lane)*16 + i]; lane: m=lane&15 (row), h=lane>>4.
// Element i -> K = ks*32 + h*16 + i
// ---------------------------------------------------------------------------
__global__ __launch_bounds__(256) void proto_packA(const float* __restrict__ emb,
                                                   __bf16* __restrict__ Apack) {
    int t = blockIdx.x * 256 + threadIdx.x;         // 1024*50*32 = 1,638,400 threads
    int lane = t & 31;
    int ks   = (t >> 5) % KST;
    int rg   = t / (32 * KST);
    int h = lane >> 4, m = lane & 15;
    const float* src = emb + (size_t)(rg * 16 + m) * E_ + ks * 32 + h * 16;
    __bf16* dst = Apack + ((size_t)(rg * KST + ks) * 32 + lane) * 16;
#pragma unroll
    for (int i = 0; i < 8; ++i) dst[i]     = (__bf16)src[i];
#pragma unroll
    for (int i = 0; i < 8; ++i) dst[8 + i] = (__bf16)src[8 + i];
}

// ---------------------------------------------------------------------------
// Pack protolayer [100,1600,4] f32 -> bf16 B-fragments, padded chunk stride.
// Chunk ks occupies CHUNKP4 uint4; element ((h*400 + c)*16 + i) within chunk
// holds proto[c>>2][ks*32 + h*16 + i][c&3].
// ---------------------------------------------------------------------------
__global__ __launch_bounds__(256) void proto_packW(const float* __restrict__ proto,
                                                   __bf16* __restrict__ Wpack) {
    int t = blockIdx.x * 256 + threadIdx.x;
    if (t >= KST * 2 * N_) return;                  // 40,000
    int c  = t % N_;
    int h  = (t / N_) & 1;
    int ks = t / (2 * N_);
    int p = c >> 2, tap = c & 3;
    int ebase = ks * 32 + h * 16;
    __bf16* dst = Wpack + (size_t)ks * (CHUNKP4 * 8) + (size_t)(h * N_ + c) * 16;
#pragma unroll
    for (int i = 0; i < 16; ++i)
        dst[i] = (__bf16)proto[(size_t)p * 6400 + (size_t)(ebase + i) * 4 + tap];
}

// s[r] = sum_e emb[r,e]^2
__global__ __launch_bounds__(128) void proto_rowss(const float* __restrict__ emb,
                                                   float* __restrict__ s) {
    int r = blockIdx.x;
    float a = 0.f;
    for (int e = threadIdx.x; e < E_; e += 128) {
        float v = emb[(size_t)r * E_ + e];
        a += v * v;
    }
    __shared__ float red[128];
    red[threadIdx.x] = a; __syncthreads();
    for (int off = 64; off > 0; off >>= 1) {
        if (threadIdx.x < off) red[threadIdx.x] += red[threadIdx.x + off];
        __syncthreads();
    }
    if (threadIdx.x == 0) s[r] = red[0];
}

// p2[p] = sum_{e,k} proto[p,e,k]^2
__global__ __launch_bounds__(256) void proto_p2(const float* __restrict__ proto,
                                                float* __restrict__ p2) {
    int p = blockIdx.x;
    float a = 0.f;
    for (int i = threadIdx.x; i < 6400; i += 256) {
        float v = proto[(size_t)p * 6400 + i];
        a += v * v;
    }
    __shared__ float red[256];
    red[threadIdx.x] = a; __syncthreads();
    for (int off = 128; off > 0; off >>= 1) {
        if (threadIdx.x < off) red[threadIdx.x] += red[threadIdx.x + off];
        __syncthreads();
    }
    if (threadIdx.x == 0) p2[p] = red[0];
}

// ---------------------------------------------------------------------------
// GEMM: G[16384 x 400] = Abf16 x Wbf16 via v_wmma_f32_16x16x32_bf16.
// Block = 128 threads (4 waves). Wave w owns rows rgroup = blockIdx.x*4+w,
// all 400 columns (25 v8f accumulators). W chunks double-buffered in LDS,
// staged with async global->LDS (ASYNCcnt); A and B fragments software-
// pipelined; one barrier per K-step.
// ---------------------------------------------------------------------------
__global__ __launch_bounds__(128) void proto_gemm(const __bf16* __restrict__ Apack,
                                                  const uint4* __restrict__ Wpack4,
                                                  float* __restrict__ G) {
    __shared__ uint4 ldsW[2 * CHUNKP4];             // 2 x 26.6 KB double buffer
    const int tid  = threadIdx.x;
    const int wave = tid >> 5;
    const int lane = tid & 31;
    const int rgroup = blockIdx.x * 4 + wave;
    const int h   = lane >> 4;
    const int nlo = lane & 15;

    // Opaque never-executed store: the async DMA writes ldsW behind the
    // compiler's back; this may-store (unfoldable predicate from asm) stops
    // LLVM from proving ldsW is never written and folding the ds_loads.
    unsigned opq;
    asm volatile("s_mov_b32 %0, 0" : "=s"(opq));
    if (opq) ldsW[(opq + tid) & 2047] = uint4{0u, 0u, 0u, 0u};

    v8f acc[NT];
#pragma unroll
    for (int i = 0; i < NT; ++i) acc[i] = v8f{};

    const uint4* Ag4 = (const uint4*)Apack;

    // --- stage chunk 0 into buffer 0: exactly 13 async b128 per thread ---
#pragma unroll
    for (int j = 0; j < 13; ++j) {
        const int e = tid + j * 128;
        async_g2l_b128((unsigned)(e * 16), Wpack4 + e);
    }
    // prime the A-fragment pipeline
    Frag a_cur;
    {
        const size_t ab = ((size_t)(rgroup * KST) * 32 + lane) * 2;
        a_cur.u[0] = Ag4[ab];
        a_cur.u[1] = Ag4[ab + 1];
    }
    wait_async0();
    __syncthreads();

    for (int ks = 0; ks < KST; ++ks) {
        const int cur = ks & 1;

        // async-stage the next chunk into the other buffer
        if (ks + 1 < KST) {
            const uint4* src = Wpack4 + (size_t)(ks + 1) * CHUNKP4;
            const int dstE = (1 - cur) * CHUNKP4;
#pragma unroll
            for (int j = 0; j < 13; ++j) {
                const int e = tid + j * 128;
                async_g2l_b128((unsigned)((dstE + e) * 16), src + e);
            }
        }
        // prefetch next A fragment (consumed next iteration)
        Frag a_nxt;
        if (ks + 1 < KST) {
            const size_t ab = ((size_t)(rgroup * KST + ks + 1) * 32 + lane) * 2;
            a_nxt.u[0] = Ag4[ab];
            a_nxt.u[1] = Ag4[ab + 1];
        }

        // B fragments, two-deep pipelined from LDS
        const uint4* wbuf = &ldsW[cur * CHUNKP4];
        const int wb0 = (h * N_ + nlo) * 2;
        Frag bb[2];
        bb[0].u[0] = wbuf[wb0];
        bb[0].u[1] = wbuf[wb0 + 1];
#pragma unroll
        for (int nt = 0; nt < NT; ++nt) {
            if (nt + 1 < NT) {
                const int wb = wb0 + (nt + 1) * 32;
                bb[(nt + 1) & 1].u[0] = wbuf[wb];
                bb[(nt + 1) & 1].u[1] = wbuf[wb + 1];
            }
            acc[nt] = __builtin_amdgcn_wmma_f32_16x16x32_bf16(
                false, a_cur.v, false, bb[nt & 1].v, (short)0, acc[nt], false, false);
        }
        if (ks + 1 < KST) a_cur = a_nxt;

        wait_async0();       // this wave's async writes to next buffer done
        __syncthreads();     // publish next buffer / retire reads of current
    }

    // C/D layout: VGPR v, lane half h -> row M = v + h*8, col N = lane&15
    const int rowbase = rgroup * 16 + h * 8;
#pragma unroll
    for (int nt = 0; nt < NT; ++nt) {
#pragma unroll
        for (int v = 0; v < 8; ++v) {
            G[(size_t)(rowbase + v) * N_ + nt * 16 + nlo] = acc[nt][v];
        }
    }
}

// ---------------------------------------------------------------------------
// Distances + per-(b,p) min.  dist = sqrt(|x2 - 2*xp + p2|),
// xp[b,p,l] = sum_k G[b*256 + l + k*d][p*4 + k]
// ---------------------------------------------------------------------------
__global__ __launch_bounds__(128) void proto_post(const float* __restrict__ G,
                                                  const float* __restrict__ s,
                                                  const float* __restrict__ p2,
                                                  float* __restrict__ out) {
    const int b = blockIdx.x / P_;
    const int p = blockIdx.x % P_;
    int d, pg, Lout, nfilt; size_t obase;
    if (p < 34)      { d = 1; pg = p;      Lout = D1LOUT; nfilt = 34; obase = OFF_D1; }
    else if (p < 67) { d = 2; pg = p - 34; Lout = D2LOUT; nfilt = 33; obase = OFF_D2; }
    else             { d = 3; pg = p - 67; Lout = D3LOUT; nfilt = 33; obase = OFF_D3; }
    const float pp = p2[p];
    float lmin = 3.4e38f;
    for (int l = threadIdx.x; l < Lout; l += 128) {
        const int rb = b * L_ + l;
        float x2 = 0.f, xp = 0.f;
#pragma unroll
        for (int k = 0; k < 4; ++k) {
            const int r = rb + k * d;
            x2 += s[r];
            xp += G[(size_t)r * N_ + p * 4 + k];
        }
        const float dist = sqrtf(fabsf(x2 - 2.f * xp + pp));
        out[obase + ((size_t)b * nfilt + pg) * Lout + l] = dist;
        lmin = fminf(lmin, dist);
    }
    __shared__ float red[128];
    red[threadIdx.x] = lmin; __syncthreads();
    for (int off = 64; off > 0; off >>= 1) {
        if (threadIdx.x < off)
            red[threadIdx.x] = fminf(red[threadIdx.x], red[threadIdx.x + off]);
        __syncthreads();
    }
    if (threadIdx.x == 0) out[(size_t)b * P_ + p] = red[0];
}

// class_out[b,c] = sum_p protodist[b,p] * fc_w[c,p]
__global__ __launch_bounds__(128) void proto_fc(const float* __restrict__ out_as_in,
                                                const float* __restrict__ fc,
                                                float* __restrict__ out) {
    const int t = threadIdx.x;      // 128 = 64 b * 2 classes
    const int b = t >> 1, c = t & 1;
    float acc = 0.f;
#pragma unroll 4
    for (int p = 0; p < P_; ++p)
        acc += out_as_in[(size_t)b * P_ + p] * fc[c * P_ + p];
    out[OFF_CL + b * 2 + c] = acc;
}

// ---------------------------------------------------------------------------
extern "C" void kernel_launch(void* const* d_in, const int* in_sizes, int n_in,
                              void* d_out, int out_size, void* d_ws, size_t ws_size,
                              hipStream_t stream) {
    const float* emb   = (const float*)d_in[0];   // [64,256,1600]
    const float* proto = (const float*)d_in[1];   // [100,1600,4]
    const float* fc    = (const float*)d_in[2];   // [2,100]
    float* out = (float*)d_out;

    // workspace carve-up
    char* ws = (char*)d_ws;
    const size_t apack_bytes = (size_t)M_ * E_ * sizeof(__bf16);        // 52,428,800
    const size_t wpack_bytes = (size_t)KST * CHUNKP4 * 16;              //  1,331,200
    const size_t g_bytes     = (size_t)M_ * N_ * sizeof(float);         // 26,214,400
    __bf16* Apack = (__bf16*)(ws);
    __bf16* Wpack = (__bf16*)(ws + apack_bytes);
    float*  G     = (float*)(ws + apack_bytes + wpack_bytes);
    float*  s     = (float*)(ws + apack_bytes + wpack_bytes + g_bytes);
    float*  p2    = (float*)(ws + apack_bytes + wpack_bytes + g_bytes + (size_t)M_ * 4 + 256);

    proto_packA<<<(M_ / 16) * KST * 32 / 256, 256, 0, stream>>>(emb, Apack);
    proto_packW<<<(KST * 2 * N_ + 255) / 256, 256, 0, stream>>>(proto, Wpack);
    proto_rowss<<<M_, 128, 0, stream>>>(emb, s);
    proto_p2<<<P_, 256, 0, stream>>>(proto, p2);
    proto_gemm<<<M_ / 64, 128, 0, stream>>>(Apack, (const uint4*)Wpack, G);
    proto_post<<<B_ * P_, 128, 0, stream>>>(G, s, p2, out);
    proto_fc<<<1, 128, 0, stream>>>(out, fc, out);
}
